// LPLayer_43379169689769
// MI455X (gfx1250) — matrix operand order repaired and checked by reading, not compiled, not run on hardware
//
#include <hip/hip_runtime.h>
#include <hip/hip_bf16.h>

// ---------------------------------------------------------------------------
// LPLayer for MI455X (gfx1250): e_x = x@W^T+b ; lp = relu(tanh(e_x@e_x^T)) ;
// node_feats = lp@e_x / N.  Big GEMMs run on v_wmma_f32_16x16x32_f16, and
// the activation uses the native CDNA5 V_TANH_F32 transcendental instead of
// the branchy libm expansion.
// B=8, N=2048, C_IN=128, C_OUT=64.
// ---------------------------------------------------------------------------

#define BB      8
#define NN      2048
#define CIN     128
#define COUT    64
#define ROWS    (BB * NN)            // 16384 global rows

typedef __attribute__((ext_vector_type(16))) _Float16 v16h;
typedef __attribute__((ext_vector_type(8)))  _Float16 v8h;
typedef __attribute__((ext_vector_type(8)))  float    v8f;
typedef __attribute__((ext_vector_type(4)))  float    v4f;

// Native CDNA5 transcendental tanh.  TRANS32 ops need one independent op or
// V_NOP before the result is consumed (ISA 01_flow_sync §7.4); the compiler
// cannot see through the asm blob, so cover the hazard inside it.
__device__ __forceinline__ float tanh_hw(float x) {
    float r;
    asm("v_tanh_f32 %0, %1\n\t"
        "v_nop"
        : "=v"(r) : "v"(x));
    return r;
}

// ---------------------------------------------------------------------------
// Kernel 1: projection e_x = x @ W^T + b  (fp32 math, f16 outputs).
// Writes e16 row-major [ROWS][COUT] and e16T [COUT][ROWS] so that both the
// Gram kernel (A/B frags from rows) and the node_feats kernel (B frag needs
// the K dimension contiguous) get coalesced 32-byte loads.
// ---------------------------------------------------------------------------
__global__ void lp_proj_kernel(const float* __restrict__ x,
                               const float* __restrict__ W,
                               const float* __restrict__ bias,
                               _Float16* __restrict__ e16,
                               _Float16* __restrict__ e16T) {
    int idx = blockIdx.x * blockDim.x + threadIdx.x;   // ROWS*COUT threads
    int row = idx >> 6;          // global row in [0, ROWS)
    int c   = idx & 63;          // output channel
    const v4f* xr = (const v4f*)(x + (long)row * CIN);
    const v4f* wr = (const v4f*)(W + (long)c   * CIN);
    float acc = bias[c];
#pragma unroll
    for (int k = 0; k < CIN / 4; ++k) {
        v4f a = xr[k], b = wr[k];
        acc += a.x * b.x + a.y * b.y + a.z * b.z + a.w * b.w;
    }
    _Float16 h = (_Float16)acc;
    e16 [(long)row * COUT + c] = h;
    e16T[(long)c * ROWS + row] = h;
}

// ---------------------------------------------------------------------------
// Kernel 2: lp = relu(tanh(e_x @ e_x^T)), one wave32 per 16x16 tile.
// K=64 -> two chained v_wmma_f32_16x16x32_f16.
// A-fragment (16-bit A 16x32): lane m holds row m%16; lanes 0-15
// carry K 0..15, lanes 16-31 carry K 16..31 (same order as B).
// B-fragment (32x16): lane n holds column n%16 (= row j of e_x); lanes 0-15
// carry K 0..15, lanes 16-31 carry K 16..31 -> one contiguous 32B load.
// ---------------------------------------------------------------------------
#define WAVES_PER_BLOCK 8

__global__ void lp_gram_kernel(const _Float16* __restrict__ e16,
                               float* __restrict__ lp) {
    const int lane = threadIdx.x & 31;
    const int wave = threadIdx.x >> 5;
    const int tile = blockIdx.x * WAVES_PER_BLOCK + wave;   // 131072 tiles
    const int b    = tile >> 14;                 // 16384 tiles per batch
    const int t    = tile & 16383;
    const int i0   = (t >> 7) << 4;              // 128 row tiles
    const int j0   = (t & 127) << 4;             // 128 col tiles

    const int m    = lane & 15;
    const int hi   = lane >> 4;                  // lane-half select
    const long rowA = (long)(b * NN + i0 + m) * COUT;
    const long rowB = (long)(b * NN + j0 + m) * COUT;

    v8f acc = {};
#pragma unroll
    for (int kb = 0; kb < COUT; kb += 32) {
        // A fragment: two 16B chunks per lane, K contiguous 16*hi..16*hi+15 (matches B)
        v8h a0 = *(const v8h*)(e16 + rowA + kb + hi * 16);
        v8h a1 = *(const v8h*)(e16 + rowA + kb + hi * 16 + 8);
        v16h afrag;
#pragma unroll
        for (int q = 0; q < 8; ++q) { afrag[q] = a0[q]; afrag[8 + q] = a1[q]; }
        // B fragment: one contiguous 32B chunk per lane
        v16h bfrag = *(const v16h*)(e16 + rowB + kb + hi * 16);
        acc = __builtin_amdgcn_wmma_f32_16x16x32_f16(
            /*neg_a=*/false, afrag, /*neg_b=*/false, bfrag,
            /*c_mod=*/(short)0, acc, /*reuse_a=*/false, /*reuse_b=*/false);
    }

    // Epilogue: relu(tanh(.)) via native v_tanh_f32, store per the C/D
    // layout (VGPR=M, lane=N).  This kernel carries the single 134MB HBM
    // store, so the epilogue must stay a handful of VALU ops per element.
    const int col   = j0 + m;
    const int rbase = i0 + hi * 8;
    float* lpb = lp + (long)b * NN * NN;
#pragma unroll
    for (int r = 0; r < 8; ++r) {
        float y = fmaxf(0.0f, tanh_hw(acc[r]));
        lpb[(long)(rbase + r) * NN + col] = y;
    }
}

// ---------------------------------------------------------------------------
// Kernel 3: node_feats = (lp @ e_x) / N.  One wave per 16x16 output tile,
// K loop over 2048 -> 64 x v_wmma_f32_16x16x32_f16.  lp (f32, hot in the
// 192MB L2) is converted to f16 on load; B comes from e16T so K is the
// contiguous axis (32B load per lane).
// ---------------------------------------------------------------------------
__global__ void lp_nodefeat_kernel(const float* __restrict__ lp,
                                   const _Float16* __restrict__ e16T,
                                   float* __restrict__ nf) {
    const int lane = threadIdx.x & 31;
    const int wave = threadIdx.x >> 5;
    const int tile = blockIdx.x * WAVES_PER_BLOCK + wave;   // 4096 tiles
    const int rb   = tile >> 2;                  // 1024 row blocks of 16
    const int ct   = tile & 3;                   // 4 col tiles of 16
    const int b    = rb >> 7;                    // 128 row blocks per batch
    const int i0   = (rb & 127) << 4;
    const int c0   = ct << 4;

    const int m  = lane & 15;
    const int hi = lane >> 4;
    const float*    lpRow = lp + ((long)b * NN + i0 + m) * NN;        // A rows
    const _Float16* bCol  = e16T + (long)(c0 + m) * ROWS + (long)b * NN; // B cols

    v8f acc = {};
#pragma unroll 4
    for (int j0 = 0; j0 < NN; j0 += 32) {
        // A fragment: lp f32 -> f16, K contiguous 16*hi..16*hi+15 as in the Gram kernel
        v8f a0 = *(const v8f*)(lpRow + j0 + hi * 16);
        v8f a1 = *(const v8f*)(lpRow + j0 + hi * 16 + 8);
        v16h afrag;
#pragma unroll
        for (int q = 0; q < 8; ++q) {
            afrag[q]     = (_Float16)a0[q];
            afrag[8 + q] = (_Float16)a1[q];
        }
        // B fragment: contiguous 16 halves along K from the transposed copy
        v16h bfrag = *(const v16h*)(bCol + j0 + hi * 16);
        acc = __builtin_amdgcn_wmma_f32_16x16x32_f16(
            false, afrag, false, bfrag, (short)0, acc, false, false);
    }

    const float scale = 1.0f / (float)NN;
    const int col   = c0 + m;
    const int rbase = i0 + hi * 8;
#pragma unroll
    for (int r = 0; r < 8; ++r) {
        nf[((long)b * NN + rbase + r) * COUT + col] = acc[r] * scale;
    }
}

// ---------------------------------------------------------------------------
// Host-side launcher.  Inputs: x [8,2048,128] f32, W_adj [64,128] f32,
// b_adj [64] f32.  Output: node_feats [8,2048,64] then lp [8,2048,2048],
// concatenated flat in d_out (both f32).  d_ws holds the 4MB f16 staging.
// ---------------------------------------------------------------------------
extern "C" void kernel_launch(void* const* d_in, const int* in_sizes, int n_in,
                              void* d_out, int out_size, void* d_ws, size_t ws_size,
                              hipStream_t stream) {
    (void)in_sizes; (void)n_in; (void)out_size; (void)ws_size;
    const float* x    = (const float*)d_in[0];
    const float* W    = (const float*)d_in[1];
    const float* bias = (const float*)d_in[2];

    float* nf = (float*)d_out;                       // [8,2048,64]
    float* lp = nf + (long)ROWS * COUT;              // [8,2048,2048]

    _Float16* e16  = (_Float16*)d_ws;                // [16384][64]
    _Float16* e16T = e16 + (long)ROWS * COUT;        // [64][16384]

    // 1) projection + f16 staging (row-major + transposed)
    {
        int threads = ROWS * COUT;                   // 1,048,576
        lp_proj_kernel<<<threads / 256, 256, 0, stream>>>(x, W, bias, e16, e16T);
    }
    // 2) Gram + tanh/relu epilogue -> lp (the single 134MB HBM store)
    {
        int tiles = BB * (NN / 16) * (NN / 16);      // 131072
        lp_gram_kernel<<<tiles / WAVES_PER_BLOCK, 32 * WAVES_PER_BLOCK, 0, stream>>>(e16, lp);
    }
    // 3) node_feats = lp @ e_x / N (lp re-read served from L2)
    {
        int tiles = (ROWS / 16) * (COUT / 16);       // 4096
        lp_nodefeat_kernel<<<tiles / WAVES_PER_BLOCK, 32 * WAVES_PER_BLOCK, 0, stream>>>(lp, e16T, nf);
    }
}
